// CAttention_20340965114206
// MI455X (gfx1250) — matrix operand order repaired and checked by reading, not compiled, not run on hardware
//
#include <hip/hip_runtime.h>
#include <hip/hip_bf16.h>

typedef __attribute__((ext_vector_type(16))) _Float16 v16h;
typedef __attribute__((ext_vector_type(8)))  _Float16 v8h;
typedef __attribute__((ext_vector_type(8)))  float    v8f;
typedef __attribute__((ext_vector_type(4)))  int      v4i;

#define Bsz 64
#define Cch 256
#define Nn  307
#define Tt  12
#define NT  (Nn * Tt)   // 3684

#if defined(__has_builtin)
#if __has_builtin(__builtin_amdgcn_global_load_async_to_lds_b128)
#define HAVE_ASYNC_LDS 1
#endif
#endif

#if HAVE_ASYNC_LDS
typedef __attribute__((address_space(1))) v4i* gv4i_p;   // global int4*
typedef __attribute__((address_space(3))) v4i* sv4i_p;   // LDS int4*
#endif

// ---------------------------------------------------------------------------
// Kernel 1: k[b,c,t] = sum_n alpha[n] * x[b,c,n,t]
// ---------------------------------------------------------------------------
__global__ void __launch_bounds__(192)
pool_kernel(const float* __restrict__ x, const float* __restrict__ alpha,
            float* __restrict__ kout) {
    const int bc = blockIdx.x;                       // 0 .. B*C-1
    const float* xp = x + (size_t)bc * (Nn * Tt);
    const int t = threadIdx.x;                       // 0..11
    const int w = threadIdx.y;                       // 0..15

    float acc = 0.f;
    for (int n = w; n < Nn; n += 16)
        acc += alpha[n] * xp[n * Tt + t];

    __shared__ float red[16][Tt];
    red[w][t] = acc;
    __syncthreads();
    if (w == 0) {
        float s = 0.f;
        #pragma unroll
        for (int j = 0; j < 16; ++j) s += red[j][t];
        kout[(size_t)bc * Tt + t] = s;
    }
}

// ---------------------------------------------------------------------------
// Kernel 2: att[b,c,:] = softmax_d( (k[b,c,:] @ W) . k[b,d,:] )
// Stores attention weights as f16 (softmax output, in [0,1]) so the WMMA
// consumer needs no A-side conversions.
// ---------------------------------------------------------------------------
__global__ void __launch_bounds__(256)
att_kernel(const float* __restrict__ kin, const float* __restrict__ W,
           _Float16* __restrict__ att) {
    const int bc = blockIdx.x;
    const int b  = bc >> 8;
    const int c  = bc & 255;
    const int d  = threadIdx.x;                      // 0..255

    __shared__ float kb[Cch * Tt];                   // 12 KB
    __shared__ float kw[Tt];
    __shared__ float red[Cch];

    const float* kbase = kin + (size_t)b * Cch * Tt;
    for (int i = threadIdx.x; i < Cch * Tt; i += 256) kb[i] = kbase[i];
    __syncthreads();

    if (threadIdx.x < Tt) {
        float s = 0.f;
        #pragma unroll
        for (int t = 0; t < Tt; ++t) s += kb[c * Tt + t] * W[t * Tt + threadIdx.x];
        kw[threadIdx.x] = s;
    }
    __syncthreads();

    float sc = 0.f;
    #pragma unroll
    for (int t = 0; t < Tt; ++t) sc += kw[t] * kb[d * Tt + t];

    red[d] = sc;
    __syncthreads();
    for (int s2 = 128; s2 > 0; s2 >>= 1) {
        if (d < s2) red[d] = fmaxf(red[d], red[d + s2]);
        __syncthreads();
    }
    const float m = red[0];
    __syncthreads();
    const float e = __expf(sc - m);
    red[d] = e;
    __syncthreads();
    for (int s2 = 128; s2 > 0; s2 >>= 1) {
        if (d < s2) red[d] += red[d + s2];
        __syncthreads();
    }
    att[(size_t)bc * Cch + d] = (_Float16)(e / red[0]);
}

// ---------------------------------------------------------------------------
// A-tile staging helper: 16 x 256 f16 tile (8 KB = 512 x 16B chunks),
// async global->LDS when available.
// ---------------------------------------------------------------------------
__device__ __forceinline__ void stage_tile(const _Float16* __restrict__ gsrc,
                                           _Float16* ldst, int tid) {
#if HAVE_ASYNC_LDS
    #pragma unroll
    for (int i = 0; i < 2; ++i) {
        const int idx = tid + i * 256;               // chunk 0..511
        __builtin_amdgcn_global_load_async_to_lds_b128(
            (gv4i_p)(gsrc + idx * 8),
            (sv4i_p)(ldst + idx * 8), 0, 0);
    }
#else
    for (int i = tid; i < 16 * Cch; i += 256) ldst[i] = gsrc[i];
#endif
}

__device__ __forceinline__ void wait_async() {
#if HAVE_ASYNC_LDS
#if __has_builtin(__builtin_amdgcn_s_wait_asynccnt)
    __builtin_amdgcn_s_wait_asynccnt(0);
#else
    asm volatile("s_wait_asynccnt 0" ::: "memory");
#endif
#endif
}

// ---------------------------------------------------------------------------
// Kernel 3: out[b] = att[b] (256x256 f16) @ X[b] (256x3684 f32->f16), f32 acc.
//
// Each wave owns one 16-wide column strip of X and holds ALL B fragments
// (K=256 -> 8 x v16h = 64 VGPRs) in registers, looping over the 16 row tiles:
// x is read from HBM exactly once. The f16 A tile (8 KB, shared by all 8
// waves) is double-buffered in LDS with async global->LDS copies overlapping
// the previous tile's WMMA work.
//
// Fragment layouts used here (wave32, contiguous k order):
//   A 16x32 f16 : lane = row(0..15) | K-half h=(lane>>4):
//       a[0..15] = A[row][k0+h*16+0..15]
//   B 32x16 f16 : lane = col(0..15) | K-group(lane>>4):
//       b[m] = B[k0 + (lane>>4)*16 + m][col]
//   C/D 16x16 f32: acc[j] -> row (lane>>4)*8 + j, col lane&15
// ---------------------------------------------------------------------------
__global__ void __launch_bounds__(256)
mix_kernel(const _Float16* __restrict__ att, const float* __restrict__ x,
           float* __restrict__ out) {
    const int b    = blockIdx.y;
    const int tcol = blockIdx.x * 8 + threadIdx.y;   // column-tile index
    const int col0 = tcol * 16;
    const int lane = threadIdx.x;                    // 0..31
    const int half = lane >> 4;
    const int l15  = lane & 15;
    const int tid  = threadIdx.y * 32 + threadIdx.x;

    const _Float16* Ab = att + (size_t)b * Cch * Cch;
    const float*    X  = x   + (size_t)b * Cch * NT;
    float*          O  = out + (size_t)b * Cch * NT;

    const int  coln  = col0 + l15;
    const bool colok = coln < NT;
    const int  colc  = colok ? coln : (NT - 1);      // clamped, in-bounds

    __shared__ alignas(16) _Float16 sA[2][16 * Cch]; // 2 x 8 KB, double buffer

    // ---- preload B fragments for the full K=256 (held in registers) ----
    v16h bf[8];
    #pragma unroll
    for (int kk = 0; kk < 8; ++kk) {
        const float* xcol = X + (size_t)(kk * 32 + half * 16) * NT + colc;
        #pragma unroll
        for (int m = 0; m < 16; ++m) {
            float v = xcol[(size_t)m * NT];          // 16-lane coalesced
            bf[kk][m] = colok ? (_Float16)v : (_Float16)0.f;
        }
    }

    // ---- prologue: stage row tile 0 ----
    stage_tile(Ab, sA[0], tid);

    for (int ct = 0; ct < 16; ++ct) {
        const int c0 = ct * 16;

        wait_async();                                // this wave's copies done
        __syncthreads();                             // all waves' copies visible

        if (ct + 1 < 16)                             // overlap: fetch next tile
            stage_tile(Ab + (size_t)(c0 + 16) * Cch, sA[(ct + 1) & 1], tid);

        const _Float16* sAt = sA[ct & 1];
        v8f acc = {};
        #pragma unroll
        for (int kk = 0; kk < 8; ++kk) {
            const _Float16* ar = sAt + l15 * Cch + kk * 32 + half * 16;
            v8h alo = *(const v8h*)(ar);             // ds_load_b128
            v8h ahi = *(const v8h*)(ar + 8);         // ds_load_b128
            v16h a;
            #pragma unroll
            for (int j = 0; j < 8; ++j) { a[j] = alo[j]; a[8 + j] = ahi[j]; }
            acc = __builtin_amdgcn_wmma_f32_16x16x32_f16(
                false, a, false, bf[kk], (short)0, acc, false, false);
        }

        if (colok) {
            #pragma unroll
            for (int j = 0; j < 8; ++j)              // row-contiguous per half
                O[(size_t)(c0 + half * 8 + j) * NT + coln] = acc[j];
        }
    }
}

// ---------------------------------------------------------------------------
extern "C" void kernel_launch(void* const* d_in, const int* in_sizes, int n_in,
                              void* d_out, int out_size, void* d_ws, size_t ws_size,
                              hipStream_t stream) {
    const float* x     = (const float*)d_in[0];
    const float* W     = (const float*)d_in[1];
    const float* alpha = (const float*)d_in[2];
    float* out = (float*)d_out;

    // workspace layout: k[B*C*T] (f32) then att[B*C*C] (f16)
    float*    kbuf = (float*)d_ws;
    _Float16* att  = (_Float16*)((char*)d_ws +
                                 (size_t)Bsz * Cch * Tt * sizeof(float));

    // 1) node pooling
    pool_kernel<<<Bsz * Cch, dim3(Tt, 16), 0, stream>>>(x, alpha, kbuf);

    // 2) bilinear scores + softmax (f16 output)
    att_kernel<<<Bsz * Cch, 256, 0, stream>>>(kbuf, W, att);

    // 3) WMMA channel mixing: 231 column tiles -> 29 blocks of 8 waves
    const int ctiles = (NT + 15) / 16;               // 231
    const int gx     = (ctiles + 7) / 8;             // 29
    mix_kernel<<<dim3(gx, Bsz), dim3(32, 8), 0, stream>>>(att, x, out);
}
